// Attention_59803124630063
// MI455X (gfx1250) — matrix operand order repaired and checked by reading, not compiled, not run on hardware
//
#include <hip/hip_runtime.h>
#include <hip/hip_bf16.h>
#include <hip/hip_fp16.h>

// ---------------------------------------------------------------------------
// CDNA5 (gfx1250) fused attention pipeline:
//   wave32 + v_wmma_f32_16x16x32_f16 + TDM (tensor_load_to_lds) double buffer.
// ---------------------------------------------------------------------------

typedef __attribute__((ext_vector_type(16))) _Float16 v16h;
typedef __attribute__((ext_vector_type(8)))  _Float16 v8h;
typedef __attribute__((ext_vector_type(8)))  float    v8f;
typedef __attribute__((ext_vector_type(4)))  unsigned int v4u;
typedef __attribute__((ext_vector_type(8)))  int      v8i_;
typedef __attribute__((ext_vector_type(4)))  int      v4i_;

#define NH_  32
#define NKV_ 8
#define HD_  64
#define NREP_ 4

#if defined(__HIP_DEVICE_COMPILE__) && __has_builtin(__builtin_amdgcn_tensor_load_to_lds)
#define USE_TDM 1
#else
#define USE_TDM 0
#endif

__device__ __forceinline__ v16h make_v16(v8h lo, v8h hh) {
  v16h r;
#pragma unroll
  for (int i = 0; i < 8; ++i) { r[i] = lo[i]; r[i + 8] = hh[i]; }
  return r;
}

// A-operand fragment (16xK row-major source row): lane holds halves
// {hi*8..+7} and {16+hi*8..+7} of its row (ISA 7.12.2).
__device__ __forceinline__ v16h frag_a(const _Float16* row, int hi) {
  return make_v16(*(const v8h*)(row + hi * 8),
                  *(const v8h*)(row + 16 + hi * 8));
}

// B-operand fragment: column n stored contiguously over K ([N][K] layout):
// lane holds halves {hi*8..+7} and {16+hi*8..+7} of its column.
__device__ __forceinline__ v16h frag_b(const _Float16* row, int hi) {
  return make_v16(*(const v8h*)(row + hi * 8),
                  *(const v8h*)(row + 16 + hi * 8));
}

__device__ __forceinline__ v8f wmma16(v16h a, v16h b, v8f c) {
  return __builtin_amdgcn_wmma_f32_16x16x32_f16(
      false, a, false, b, (short)0, c, false, false);
}

#if USE_TDM
// 32-bit LDS offset of a __shared__ object (generic -> addrspace(3) -> int).
__device__ __forceinline__ unsigned int lds_off_u32(const void* p) {
  return (unsigned int)(uintptr_t)(__attribute__((address_space(3))) const char*)p;
}

// TDM: DMA a [tile_d1 x tile_d0] f16 tile (row stride = tensor elements K)
// from global into LDS, padding each 64B row with 16B (-> 40-half LDS stride).
// D# layout per CDNA5 ISA 8.3/8.4; groups 2/3 zero (2D tensor).
// This toolchain's builtin takes 6 operands (g0, g1, g2, g3, extra, cpol).
__device__ __forceinline__ void tdm_load_tile(unsigned int lds_addr,
                                              const void* gaddr,
                                              unsigned int Kdim) {
  const unsigned long long ga = (unsigned long long)(uintptr_t)gaddr;
  v4u g0;
  g0[0] = 1u;                                   // count=1 (valid), user mode
  g0[1] = lds_addr;                             // lds_addr
  g0[2] = (unsigned int)(ga & 0xffffffffull);   // global_addr[31:0]
  g0[3] = (unsigned int)((ga >> 32) & 0x1ffffffull) | (2u << 30);  // type=2
  v8i_ g1;
  // data_size=2B (code1) | pad_enable | pad_interval=16DW (code3, bits24:22)
  // | pad_amount=4DW (code3, bits31:25)
  g1[0] = (int)((1u << 16) | (1u << 20) | (3u << 22) | (3u << 25));
  g1[1] = (int)((Kdim & 0xffffu) << 16);                    // tensor_dim0 lo16
  g1[2] = (int)(((Kdim >> 16) & 0xffffu) | (128u << 16));   // dim0 hi / dim1=128
  g1[3] = (int)(32u << 16);                                 // tile_dim0=32
  g1[4] = (int)(128u);                                      // tile_dim1=128
  g1[5] = (int)Kdim;                                        // dim0_stride lo32
  g1[6] = 0;
  g1[7] = 0;
  const v4i_ z4 = {0, 0, 0, 0};
  const v8i_ z8 = {0, 0, 0, 0, 0, 0, 0, 0};
  __builtin_amdgcn_tensor_load_to_lds(g0, g1, z4, z4, z8, 0);
}
#endif

// ---------------------------------------------------------------------------
// f32 -> f16 cast
// ---------------------------------------------------------------------------
__global__ __launch_bounds__(256) void cast_kernel(const float* __restrict__ src,
                                                   _Float16* __restrict__ dst,
                                                   int n) {
  int i = blockIdx.x * blockDim.x + threadIdx.x;
  if (i < n) dst[i] = (_Float16)src[i];
}

// ---------------------------------------------------------------------------
// C[M,N] (f32) = A[M,K] (f16) * B[N,K]^T (f16).
// 256 threads = 8 waves; block tile 128x128, K-step 32; wave tile 64x32.
// Double-buffered LDS; tiles fetched by the Tensor Data Mover when available.
// ---------------------------------------------------------------------------
__global__ __launch_bounds__(256) void gemm_f16f32(
    const _Float16* __restrict__ A, const _Float16* __restrict__ Bw,
    float* __restrict__ C, int M, int N, int K) {
  __shared__ __align__(16) _Float16 As[2][128][40];  // 80B rows (64B + 16B pad)
  __shared__ __align__(16) _Float16 Bs[2][128][40];

  const int tid  = threadIdx.x;
  const int lane = tid & 31;
  const int wid  = tid >> 5;
  const int wm   = wid & 1;   // 2 wave rows  x 64
  const int wn   = wid >> 1;  // 4 wave cols  x 32
  const int bm   = blockIdx.y * 128;
  const int bn   = blockIdx.x * 128;
  const int r    = lane & 15;
  const int hi   = lane >> 4;

  v8f acc[4][2];
#pragma unroll
  for (int i = 0; i < 4; ++i)
#pragma unroll
    for (int j = 0; j < 2; ++j) acc[i][j] = (v8f){};

  const int nsteps = K >> 5;

#if USE_TDM
  const unsigned int aoff[2] = {lds_off_u32(&As[0][0][0]), lds_off_u32(&As[1][0][0])};
  const unsigned int boff[2] = {lds_off_u32(&Bs[0][0][0]), lds_off_u32(&Bs[1][0][0])};
  if (tid < 32) {  // wave 0 drives the TDM (EXEC-independent, one op per wave)
    tdm_load_tile(aoff[0], A + (size_t)bm * K, (unsigned)K);
    tdm_load_tile(boff[0], Bw + (size_t)bn * K, (unsigned)K);
  }
#else
  const int lr = tid >> 1;
  const int lc = (tid & 1) * 16;
#endif

  for (int step = 0; step < nsteps; ++step) {
    const int buf = step & 1;
#if USE_TDM
    if (tid < 32) {
      if (step + 1 < nsteps) {  // prefetch next tile pair into the other buffer
        tdm_load_tile(aoff[buf ^ 1], A + (size_t)bm * K + (size_t)(step + 1) * 32,
                      (unsigned)K);
        tdm_load_tile(boff[buf ^ 1], Bw + (size_t)bn * K + (size_t)(step + 1) * 32,
                      (unsigned)K);
        __builtin_amdgcn_s_wait_tensorcnt(2);  // in-order: current pair done
      } else {
        __builtin_amdgcn_s_wait_tensorcnt(0);
      }
    }
#else
    {
      const int k0 = step * 32;
      const _Float16* ga = A  + (size_t)(bm + lr) * K + k0 + lc;
      const _Float16* gb = Bw + (size_t)(bn + lr) * K + k0 + lc;
      v8h a0 = *(const v8h*)ga, a1 = *(const v8h*)(ga + 8);
      v8h b0 = *(const v8h*)gb, b1 = *(const v8h*)(gb + 8);
      *(v8h*)&As[buf][lr][lc] = a0; *(v8h*)&As[buf][lr][lc + 8] = a1;
      *(v8h*)&Bs[buf][lr][lc] = b0; *(v8h*)&Bs[buf][lr][lc + 8] = b1;
    }
#endif
    __syncthreads();  // tile visible to all 8 waves

    v16h afr[4], bfr[2];
#pragma unroll
    for (int i = 0; i < 4; ++i)
      afr[i] = frag_a(&As[buf][wm * 64 + i * 16 + r][0], hi);
#pragma unroll
    for (int j = 0; j < 2; ++j)
      bfr[j] = frag_b(&Bs[buf][wn * 32 + j * 16 + r][0], hi);
#pragma unroll
    for (int i = 0; i < 4; ++i)
#pragma unroll
      for (int j = 0; j < 2; ++j) acc[i][j] = wmma16(afr[i], bfr[j], acc[i][j]);

    __syncthreads();  // all reads done before this buffer is refilled
  }

#pragma unroll
  for (int i = 0; i < 4; ++i)
#pragma unroll
    for (int j = 0; j < 2; ++j) {
      float* cp = C + (size_t)(bm + wm * 64 + i * 16 + hi * 8) * N +
                  (bn + wn * 32 + j * 16 + r);
#pragma unroll
      for (int e = 0; e < 8; ++e) cp[(size_t)e * N] = acc[i][j][e];
    }
}

// ---------------------------------------------------------------------------
// RoPE + scatter f32 QKV rows [M,3072] into head-major f16 Q/K/V.
// grid (S, 48, B), block 64 (hs<32: Q heads, <40: K heads, else V heads).
// ---------------------------------------------------------------------------
__global__ __launch_bounds__(64) void rope_scatter(
    const float* __restrict__ qkv, const float* __restrict__ rc,
    const float* __restrict__ rs, _Float16* __restrict__ Qh,
    _Float16* __restrict__ Kh, _Float16* __restrict__ Vh, int S) {
  const int d  = threadIdx.x;
  const int s  = blockIdx.x;
  const int hs = blockIdx.y;
  const int b  = blockIdx.z;
  const size_t row = ((size_t)b * S + s) * 3072;
  const int d2 = d & 31;
  const float c  = rc[s * 32 + d2];
  const float sn = rs[s * 32 + d2];

  if (hs < NH_) {
    const int col = hs * 64;
    float v  = qkv[row + col + d];
    float ov = qkv[row + col + (d < 32 ? d + 32 : d - 32)];
    float out = (d < 32) ? (v * c - ov * sn) : (v * c + ov * sn);
    Qh[((size_t)(b * NH_ + hs) * S + s) * HD_ + d] = (_Float16)out;
  } else if (hs < NH_ + NKV_) {
    const int h = hs - NH_, col = 2048 + h * 64;
    float v  = qkv[row + col + d];
    float ov = qkv[row + col + (d < 32 ? d + 32 : d - 32)];
    float out = (d < 32) ? (v * c - ov * sn) : (v * c + ov * sn);
    Kh[((size_t)(b * NKV_ + h) * S + s) * HD_ + d] = (_Float16)out;
  } else {
    const int h = hs - NH_ - NKV_, col = 2560 + h * 64;
    Vh[((size_t)(b * NKV_ + h) * S + s) * HD_ + d] =
        (_Float16)qkv[row + col + d];
  }
}

// ---------------------------------------------------------------------------
// Flash attention with doc-mask + causal mask.
// grid (S/64, NH, B); 128 threads = 4 waves; each wave owns 16 q-rows.
// ---------------------------------------------------------------------------
__global__ __launch_bounds__(128) void attn_kernel(
    const _Float16* __restrict__ Qh, const _Float16* __restrict__ Kh,
    const _Float16* __restrict__ Vh, const int* __restrict__ doc_ids,
    _Float16* __restrict__ Oh, int S) {
  __shared__ __align__(16) _Float16 Kt[64][72];       // [krow][d]
  __shared__ __align__(16) _Float16 Vt[64][72];       // transposed: [d][krow]
  __shared__ __align__(16) float    Sc[4][16][68];    // per-wave scores
  __shared__ __align__(16) _Float16 Pt[4][16][72];    // per-wave P (f16)
  __shared__ float Mrow[4][16], Lrow[4][16], Arow[4][16];
  __shared__ int   kdoc[64];

  const int tid  = threadIdx.x;
  const int lane = tid & 31;
  const int wid  = tid >> 5;
  const int r    = lane & 15;
  const int hi   = lane >> 4;
  const int b    = blockIdx.z;
  const int h    = blockIdx.y;
  const int qt   = blockIdx.x;
  const int kvh  = h / NREP_;
  const int qbase = qt * 64;
  const int qrow0 = qbase + wid * 16;

  // Preload Q A-fragments; softmax scale 1/sqrt(64)=0.125 folded in (exact).
  const _Float16* qp = Qh + ((size_t)(b * NH_ + h) * S + qrow0 + r) * HD_;
  v16h aq[2];
#pragma unroll
  for (int kc = 0; kc < 2; ++kc) {
    v16h q = make_v16(*(const v8h*)(qp + kc * 32 + hi * 8),
                      *(const v8h*)(qp + kc * 32 + 16 + hi * 8));
#pragma unroll
    for (int i = 0; i < 16; ++i) q[i] = q[i] * (_Float16)0.125f;
    aq[kc] = q;
  }

  v8f o[4];
#pragma unroll
  for (int j = 0; j < 4; ++j) o[j] = (v8f){};

  if (lane < 16) { Mrow[wid][lane] = -3.0e38f; Lrow[wid][lane] = 0.0f; }
  int qd = 0, qpos = 0;
  if (lane < 16) {
    qpos = qrow0 + lane;
    qd   = doc_ids[(size_t)b * S + qpos];
  }

  for (int t = 0; t <= qt; ++t) {
    const int kbase = t * 64;
    __syncthreads();
    {  // cooperative K/V tile load; V stored transposed for B-fragments
      const int krow = tid >> 1;
      const int dh   = (tid & 1) * 32;
      const size_t kvrow = ((size_t)(b * NKV_ + kvh) * S + kbase + krow) * HD_;
      const _Float16* kp = Kh + kvrow + dh;
      const _Float16* vp = Vh + kvrow + dh;
      v8h k0 = *(const v8h*)(kp),      k1 = *(const v8h*)(kp + 8);
      v8h k2 = *(const v8h*)(kp + 16), k3 = *(const v8h*)(kp + 24);
      *(v8h*)&Kt[krow][dh]      = k0; *(v8h*)&Kt[krow][dh + 8]  = k1;
      *(v8h*)&Kt[krow][dh + 16] = k2; *(v8h*)&Kt[krow][dh + 24] = k3;
      v8h v0 = *(const v8h*)(vp),      v1 = *(const v8h*)(vp + 8);
      v8h v2 = *(const v8h*)(vp + 16), v3 = *(const v8h*)(vp + 24);
#pragma unroll
      for (int i = 0; i < 8; ++i) {
        Vt[dh + i][krow]      = v0[i];
        Vt[dh + 8 + i][krow]  = v1[i];
        Vt[dh + 16 + i][krow] = v2[i];
        Vt[dh + 24 + i][krow] = v3[i];
      }
      if (tid < 64) kdoc[tid] = doc_ids[(size_t)b * S + kbase + tid];
    }
    __syncthreads();

    // S = Q @ K^T (16x64 per wave)
#pragma unroll
    for (int j = 0; j < 4; ++j) {
      v8f s = (v8f){};
      s = wmma16(aq[0], frag_b(&Kt[j * 16 + r][0], hi), s);
      s = wmma16(aq[1], frag_b(&Kt[j * 16 + r][0] + 32, hi), s);
#pragma unroll
      for (int e = 0; e < 8; ++e) Sc[wid][hi * 8 + e][j * 16 + r] = s[e];
    }
    __syncthreads();

    // Online softmax: lane < 16 owns one q-row
    if (lane < 16) {
      const int row = lane;
      float mold = Mrow[wid][row];
      float tmax = -3.0e38f;
#pragma unroll 4
      for (int c = 0; c < 64; ++c) {
        bool ok = ((kbase + c) <= qpos) && (kdoc[c] == qd);
        tmax = fmaxf(tmax, ok ? Sc[wid][row][c] : -3.0e38f);
      }
      float mnew  = fmaxf(mold, tmax);
      float alpha = __expf(mold - mnew);
      float sum = 0.0f;
#pragma unroll 4
      for (int c = 0; c < 64; ++c) {
        bool ok = ((kbase + c) <= qpos) && (kdoc[c] == qd);
        float p = ok ? __expf(Sc[wid][row][c] - mnew) : 0.0f;
        sum += p;
        Pt[wid][row][c] = (_Float16)p;
      }
      Mrow[wid][row] = mnew;
      Lrow[wid][row] = Lrow[wid][row] * alpha + sum;
      Arow[wid][row] = alpha;
    }
    __syncthreads();

    // Rescale O and accumulate P @ V
    float av[8];
#pragma unroll
    for (int e = 0; e < 8; ++e) av[e] = Arow[wid][hi * 8 + e];
    v16h p0 = frag_a(&Pt[wid][r][0], hi);
    v16h p1 = frag_a(&Pt[wid][r][0] + 32, hi);
#pragma unroll
    for (int j = 0; j < 4; ++j) {
#pragma unroll
      for (int e = 0; e < 8; ++e) o[j][e] *= av[e];
      o[j] = wmma16(p0, frag_b(&Vt[j * 16 + r][0], hi), o[j]);
      o[j] = wmma16(p1, frag_b(&Vt[j * 16 + r][0] + 32, hi), o[j]);
    }
  }

  // Normalize and store to [B,S,NH*HD] f16 (A-matrix of output GEMM)
#pragma unroll
  for (int j = 0; j < 4; ++j)
#pragma unroll
    for (int e = 0; e < 8; ++e) {
      const int row = hi * 8 + e;
      const float l = Lrow[wid][row];
      const int s = qbase + wid * 16 + row;
      Oh[((size_t)(b * S + s)) * (NH_ * HD_) + h * HD_ + j * 16 + r] =
          (_Float16)(o[j][e] / l);
    }
}

// ---------------------------------------------------------------------------
extern "C" void kernel_launch(void* const* d_in, const int* in_sizes, int n_in,
                              void* d_out, int out_size, void* d_ws,
                              size_t ws_size, hipStream_t stream) {
  (void)in_sizes; (void)n_in; (void)out_size; (void)ws_size;
  const float* x   = (const float*)d_in[0];
  const float* rc  = (const float*)d_in[1];
  const float* rs  = (const float*)d_in[2];
  const int*   doc = (const int*)d_in[3];
  const float* Wq  = (const float*)d_in[4];
  const float* Wk  = (const float*)d_in[5];
  const float* Wv  = (const float*)d_in[6];
  const float* Wo  = (const float*)d_in[7];

  constexpr int Bv = 2, Sv = 2048, D = 2048;
  constexpr int M = Bv * Sv;  // 4096

  char* ws = (char*)d_ws;
  size_t off = 0;
  auto alloc = [&](size_t bytes) -> void* {
    void* p = ws + off;
    off = (off + bytes + 255) & ~(size_t)255;
    return p;
  };
  _Float16* xh   = (_Float16*)alloc((size_t)M * D * 2);
  _Float16* wqkv = (_Float16*)alloc((size_t)3072 * D * 2);
  _Float16* woh  = (_Float16*)alloc((size_t)D * D * 2);
  float*    qkv  = (float*)alloc((size_t)M * 3072 * 4);
  _Float16* Qh   = (_Float16*)alloc((size_t)Bv * NH_ * Sv * HD_ * 2);
  _Float16* Kh   = (_Float16*)alloc((size_t)Bv * NKV_ * Sv * HD_ * 2);
  _Float16* Vh   = (_Float16*)alloc((size_t)Bv * NKV_ * Sv * HD_ * 2);
  _Float16* Ah   = (_Float16*)alloc((size_t)M * D * 2);

  auto cast = [&](const float* s, _Float16* d, int n) {
    cast_kernel<<<(n + 255) / 256, 256, 0, stream>>>(s, d, n);
  };
  cast(x,  xh,   M * D);
  cast(Wq, wqkv,             2048 * D);
  cast(Wk, wqkv + 2048 * D,   512 * D);
  cast(Wv, wqkv + 2560 * D,   512 * D);
  cast(Wo, woh,              2048 * D);

  // QKV projection: [4096,2048] x [3072,2048]^T
  gemm_f16f32<<<dim3(3072 / 128, M / 128), 256, 0, stream>>>(
      xh, wqkv, qkv, M, 3072, D);

  // RoPE + head-major scatter
  rope_scatter<<<dim3(Sv, NH_ + 2 * NKV_, Bv), 64, 0, stream>>>(
      qkv, rc, rs, Qh, Kh, Vh, Sv);

  // Flash attention
  attn_kernel<<<dim3(Sv / 64, NH_, Bv), 128, 0, stream>>>(
      Qh, Kh, Vh, doc, Ah, Sv);

  // Output projection: [4096,2048] x [2048,2048]^T -> f32 d_out
  gemm_f16f32<<<dim3(D / 128, M / 128), 256, 0, stream>>>(
      Ah, woh, (float*)d_out, M, D, D);
}
